// DeepseekV2Gate_CPP_44848048505223
// MI455X (gfx1250) — matrix operand order repaired and checked by reading, not compiled, not run on hardware
//
#include <hip/hip_runtime.h>
#include <stdint.h>

typedef __bf16 bf16;
typedef __attribute__((ext_vector_type(16))) __bf16 v16bf;
typedef __attribute__((ext_vector_type(8)))  __bf16 v8bf;
typedef __attribute__((ext_vector_type(8)))  float  v8f;
typedef unsigned int u32x4 __attribute__((ext_vector_type(4)));
typedef int          i32x4 __attribute__((ext_vector_type(4)));
typedef int          i32x8 __attribute__((ext_vector_type(8)));

#if __has_builtin(__builtin_amdgcn_tensor_load_to_lds) && \
    __has_builtin(__builtin_amdgcn_s_wait_tensorcnt)
#define HAVE_TDM 1
#else
#define HAVE_TDM 0
#endif

#define HIDDEN   7168
#define NEXP     256
#define KC       64            // K-chunk staged in LDS per iteration
#define NCHUNK   (HIDDEN / KC) // 112
#define TOKTILE  64            // tokens per workgroup
#define NTHREADS 512           // 16 waves
#define NGROUP   8
#define EPG      32
#define TOPK     8
#define TOPKG    4

__device__ __forceinline__ v8bf cvt8(float4 a, float4 b) {
  v8bf r;
  r[0] = (bf16)a.x; r[1] = (bf16)a.y; r[2] = (bf16)a.z; r[3] = (bf16)a.w;
  r[4] = (bf16)b.x; r[5] = (bf16)b.y; r[6] = (bf16)b.z; r[7] = (bf16)b.w;
  return r;
}

__device__ __forceinline__ float sigmoidf_fast(float x) {
  return 1.0f / (1.0f + __expf(-x));
}

// ---------------- parallel routing epilogue (8 threads per token) ----------
// smem layout (ROWS = tokens resident per phase):
//   sScore [ROWS][257] f32   : unbiased sigmoid scores
//   sGsc   [ROWS][8]   f32   : group scores
//   sSelG  [ROWS][4]   i32   : selected group ids
//   sCV/sCI[ROWS][4][8]      : per-selected-group sorted top-8 candidates
template <int ROWS>
__device__ __forceinline__ void epilogue(unsigned char* smem,
    const v8f (&acc)[4], int tokTile, int exBase, int l16, int lhalf, int tid,
    const float* sBias, int* __restrict__ outIdx, float* __restrict__ outW,
    int tok0) {
  float (*sScore)[257] = (float(*)[257])smem;
  float (*sGsc)[NGROUP] = (float(*)[NGROUP])(smem + ROWS * 1028);
  int   (*sSelG)[TOPKG] = (int(*)[TOPKG])(smem + ROWS * (1028 + 32));
  float (*sCV)[TOPKG][TOPK] = (float(*)[TOPKG][TOPK])(smem + ROWS * (1028 + 32 + 16));
  int   (*sCI)[TOPKG][TOPK] = (int(*)[TOPKG][TOPK])(smem + ROWS * (1028 + 32 + 16 + 128));

  const int NPH  = TOKTILE / ROWS;   // 1 (TDM kernel) or 2 (fallback)
  const int tokE = tid >> 3;         // token slot 0..63
  const int grpE = tid & 7;          // expert group 0..7

  for (int ph = 0; ph < NPH; ++ph) {
    __syncthreads();                 // GEMM done / previous phase readers done
    if (NPH == 1 || (tokTile >> 1) == ph) {
      const int rbase = (NPH == 1 ? tokTile : (tokTile & 1)) * 16;
#pragma unroll
      for (int t = 0; t < 4; ++t)
#pragma unroll
        for (int r = 0; r < 8; ++r)   // 512-way parallel sigmoid
          sScore[rbase + r + 8 * lhalf][exBase + t * 16 + l16] =
              sigmoidf_fast(acc[t][r]);
    }
    __syncthreads();
    if (tokE < ROWS) {               // group score = top-2 sum, rotated scan
      float m1 = -3.0e38f, m2 = -3.0e38f;
      for (int j = 0; j < EPG; ++j) {
        const int e = grpE * EPG + ((j + grpE * 4) & 31);
        const float b = sScore[tokE][e] + sBias[e];
        if (b > m1) { m2 = m1; m1 = b; }
        else if (b > m2) { m2 = b; }
      }
      sGsc[tokE][grpE] = m1 + m2;
    }
    __syncthreads();
    if (tokE < ROWS && grpE == 0) {  // leader: stable top-4 groups
      unsigned gmask = 0;
      for (int s = 0; s < TOPKG; ++s) {
        int best = 0; float bv = -3.0e38f;
        for (int g = 0; g < NGROUP; ++g)
          if (!((gmask >> g) & 1u) && sGsc[tokE][g] > bv) {
            bv = sGsc[tokE][g]; best = g;
          }
        gmask |= 1u << best;
        sSelG[tokE][s] = best;
      }
    }
    __syncthreads();
    if (tokE < ROWS) {               // selected groups: sorted top-8 lists
      int slot = -1;
#pragma unroll
      for (int s = 0; s < TOPKG; ++s)
        if (sSelG[tokE][s] == grpE) slot = s;
      if (slot >= 0) {
        float tv[TOPK]; int ti[TOPK];
#pragma unroll
        for (int k = 0; k < TOPK; ++k) { tv[k] = -3.0e38f; ti[k] = NEXP; }
        for (int j = 0; j < EPG; ++j) {
          const int e = grpE * EPG + ((j + grpE * 4) & 31);
          const float b = sScore[tokE][e] + sBias[e];
          // lexicographic (value desc, index asc) -> scan-order independent
          if (b > tv[TOPK - 1] || (b == tv[TOPK - 1] && e < ti[TOPK - 1])) {
            int q = TOPK - 1;
            while (q > 0 && (b > tv[q - 1] || (b == tv[q - 1] && e < ti[q - 1]))) {
              tv[q] = tv[q - 1]; ti[q] = ti[q - 1]; --q;
            }
            tv[q] = b; ti[q] = e;
          }
        }
#pragma unroll
        for (int k = 0; k < TOPK; ++k) {
          sCV[tokE][slot][k] = tv[k];
          sCI[tokE][slot][k] = ti[k];
        }
      }
    }
    __syncthreads();
    if (tokE < ROWS && grpE == 0) {  // leader: 4-way merge of sorted lists
      int head[TOPKG] = {0, 0, 0, 0};
      float w[TOPK]; int idx[TOPK]; float wsum = 0.0f;
#pragma unroll
      for (int k = 0; k < TOPK; ++k) {
        int bs = 0; float bv = -3.0e38f; int bi = NEXP;
#pragma unroll
        for (int s = 0; s < TOPKG; ++s) {
          if (head[s] < TOPK) {
            const float v = sCV[tokE][s][head[s]];
            const int   i = sCI[tokE][s][head[s]];
            if (v > bv || (v == bv && i < bi)) { bv = v; bi = i; bs = s; }
          }
        }
        head[bs]++;
        idx[k] = bi;
        w[k]   = sScore[tokE][bi];   // weight from UNbiased sigmoid
        wsum  += w[k];
      }
      const float scale = 2.5f / (wsum + 1e-20f);
      const size_t o = (size_t)(tok0 + ph * ROWS + tokE) * TOPK;
#pragma unroll
      for (int k = 0; k < TOPK; ++k) {
        outIdx[o + k] = idx[k];
        outW[o + k]   = w[k] * scale;
      }
    }
  }
}

// One-time weight f32 -> bf16 pre-pass into workspace.
__global__ __launch_bounds__(256) void convert_weight_kernel(
    const float* __restrict__ w, bf16* __restrict__ wb) {
  const size_t i = (size_t)blockIdx.x * blockDim.x + threadIdx.x; // per 8 elems
  const float4* src = (const float4*)w + i * 2;
  float4 f0 = src[0], f1 = src[1];
  *(v8bf*)(wb + i * 8) = cvt8(f0, f1);
}

#if HAVE_TDM
// =====================  TDM + double-buffered pipeline  =====================
__global__ __launch_bounds__(NTHREADS) void deepseek_gate_kernel_tdm(
    const float* __restrict__ hidden, const bf16* __restrict__ weightb,
    const float* __restrict__ bias, int* __restrict__ outIdx,
    float* __restrict__ outW) {
  // A: 2 x 64 x 72 bf16 = 18432 B ; B: 2 x 256 x 72 bf16 = 73728 B -> 92160 B
  // Epilogue reuse needs 64 * 1332 = 85248 B.
  __shared__ __align__(16) unsigned char smem[2 * 64 * 72 * 2 + 2 * 256 * 72 * 2];
  __shared__ float sBias[NEXP];
  bf16 (*sA)[64][72]  = (bf16(*)[64][72])smem;
  bf16 (*sB)[256][72] = (bf16(*)[256][72])(smem + 2 * 64 * 72 * 2);

  const int tid     = threadIdx.x;
  const int lane    = tid & 31;
  const int wave    = tid >> 5;          // 0..15
  const int l16     = lane & 15;
  const int lhalf   = lane >> 4;
  const int tokTile = wave >> 2;         // 0..3
  const int exBase  = (wave & 3) * 64;
  const int tok0    = blockIdx.x * TOKTILE;
  const int ldr     = tid >> 3;          // 0..63
  const int ldc     = (tid & 7) * 8;     // 0..56

  if (tid < NEXP) sBias[tid] = bias[tid];

  float4 a0, a1;                          // A staging registers
  auto load_A = [&](int c) {
    const float* asrc = hidden + (size_t)(tok0 + ldr) * HIDDEN + c * KC + ldc;
    a0 = ((const float4*)asrc)[0];
    a1 = ((const float4*)asrc)[1];
    if (c + 2 < NCHUNK) __builtin_prefetch(asrc + 2 * KC, 0, 3);
  };
  auto store_A = [&](int buf) { *(v8bf*)&sA[buf][ldr][ldc] = cvt8(a0, a1); };

  // TDM D# group1 (constant): bf16 2D tensor, tile 64 x 256 rows,
  // row stride HIDDEN elements, LDS pad 4 dwords per 32 dwords -> 72-elem rows.
  i32x8 g1;
  g1[0] = (1 << 16)       // data_size = 2 bytes
        | (1 << 20)       // pad_enable
        | (4 << 22)       // pad_interval: 32 dwords (enc 4)
        | (3 << 25);      // pad_amount : 4 dwords  (enc 3)
  g1[1] = (HIDDEN & 0xFFFF) << 16;                   // tensor_dim0[15:0]
  g1[2] = (HIDDEN >> 16) | ((NEXP & 0xFFFF) << 16);  // dim0[31:16] | dim1[15:0]
  g1[3] = (NEXP >> 16) | (KC << 16);                 // dim1[31:16] | tile_dim0
  g1[4] = NEXP;                                      // tile_dim1 (tile_dim2=0)
  g1[5] = HIDDEN;                                    // tensor_dim0_stride[31:0]
  g1[6] = 0;
  g1[7] = 0;

  auto tdm_issue = [&](int c, int buf) {
    const uint64_t gaddr = (uint64_t)(uintptr_t)(weightb + (size_t)c * KC);
    const unsigned laddr = (unsigned)(uintptr_t)&sB[buf][0][0]; // LDS byte offset
    u32x4 g0;
    g0[0] = 1u;                                       // count = 1
    g0[1] = laddr;                                    // lds_addr
    g0[2] = (unsigned)gaddr;                          // global_addr[31:0]
    g0[3] = (unsigned)(gaddr >> 32) | (2u << 30);     // addr[56:32] | type=2
    i32x4 gz = {0, 0, 0, 0};
#if defined(__clang_major__) && (__clang_major__ >= 23)
    i32x8 gz8 = {0, 0, 0, 0, 0, 0, 0, 0};
    __builtin_amdgcn_tensor_load_to_lds(g0, g1, gz, gz, gz8, 0);
#else
    __builtin_amdgcn_tensor_load_to_lds(g0, g1, gz, gz, 0);
#endif
  };

  v8f acc[4] = {};

  // prologue: fill buffer 0, start A loads for chunk 1
  load_A(0);
  store_A(0);
  if (wave == 0) tdm_issue(0, 0);
  load_A(1);

  for (int c = 0; c < NCHUNK; ++c) {
    const int buf = c & 1;
    if (wave == 0) __builtin_amdgcn_s_wait_tensorcnt(0); // B(c) landed in LDS
    __syncthreads();  // publishes sA[buf], sB[buf]; compute c-1 done everywhere
    if (c + 1 < NCHUNK) {
      store_A(buf ^ 1);                        // overlaps compute below
      if (wave == 0) tdm_issue(c + 1, buf ^ 1); // DMA overlaps compute below
      if (c + 2 < NCHUNK) load_A(c + 2);       // globals in flight during compute
    }
#pragma unroll
    for (int ks = 0; ks < 2; ++ks) {
      const int ka = ks * 32;
      const int arow = tokTile * 16 + l16;
      v8bf alo = *(const v8bf*)&sA[buf][arow][ka + lhalf * 8];
      v8bf ahi = *(const v8bf*)&sA[buf][arow][ka + 16 + lhalf * 8];
      v16bf afrag = __builtin_shufflevector(alo, ahi,
          0,1,2,3,4,5,6,7,8,9,10,11,12,13,14,15);
#pragma unroll
      for (int t = 0; t < 4; ++t) {
        const int brow = exBase + t * 16 + l16;
        v8bf blo = *(const v8bf*)&sB[buf][brow][ka + lhalf * 8];
        v8bf bhi = *(const v8bf*)&sB[buf][brow][ka + 16 + lhalf * 8];
        v16bf bfrag = __builtin_shufflevector(blo, bhi,
            0,1,2,3,4,5,6,7,8,9,10,11,12,13,14,15);
        acc[t] = __builtin_amdgcn_wmma_f32_16x16x32_bf16(
            false, afrag, false, bfrag, (short)0, acc[t], false, false);
      }
    }
  }

  epilogue<64>(smem, acc, tokTile, exBase, l16, lhalf, tid, sBias,
               outIdx, outW, tok0);
}
#endif  // HAVE_TDM

// ============  fallback: cooperative-load pipeline (round-2 style) ==========
template <bool PRE>
__global__ __launch_bounds__(NTHREADS) void deepseek_gate_kernel(
    const float* __restrict__ hidden, const float* __restrict__ weightf,
    const bf16* __restrict__ weightb, const float* __restrict__ bias,
    int* __restrict__ outIdx, float* __restrict__ outW) {
  __shared__ __align__(16) unsigned char smem[46080];
  __shared__ float sBias[NEXP];
  bf16 (*sA)[72] = (bf16(*)[72])smem;
  bf16 (*sB)[72] = (bf16(*)[72])(smem + 64 * 72 * 2);

  const int tid     = threadIdx.x;
  const int lane    = tid & 31;
  const int wave    = tid >> 5;
  const int l16     = lane & 15;
  const int lhalf   = lane >> 4;
  const int tokTile = wave >> 2;
  const int exBase  = (wave & 3) * 64;
  const int tok0    = blockIdx.x * TOKTILE;
  const int ldr     = tid >> 3;
  const int ldc     = (tid & 7) * 8;

  if (tid < NEXP) sBias[tid] = bias[tid];

  float4 a0, a1;
  v8bf   bs[4];
  float4 bf0[4], bf1[4];

  auto load_stage = [&](int c) {
    const int k0 = c * KC;
    const float* asrc = hidden + (size_t)(tok0 + ldr) * HIDDEN + k0 + ldc;
    a0 = ((const float4*)asrc)[0];
    a1 = ((const float4*)asrc)[1];
    if (c + 2 < NCHUNK) __builtin_prefetch(asrc + 2 * KC, 0, 3);
    if constexpr (PRE) {
#pragma unroll
      for (int p = 0; p < 4; ++p) {
        const bf16* bsrc = weightb + (size_t)(p * 64 + ldr) * HIDDEN + k0 + ldc;
        bs[p] = *(const v8bf*)bsrc;
      }
    } else {
#pragma unroll
      for (int p = 0; p < 4; ++p) {
        const float* bsrc = weightf + (size_t)(p * 64 + ldr) * HIDDEN + k0 + ldc;
        bf0[p] = ((const float4*)bsrc)[0];
        bf1[p] = ((const float4*)bsrc)[1];
      }
    }
  };
  auto store_stage = [&]() {
    *(v8bf*)&sA[ldr][ldc] = cvt8(a0, a1);
    if constexpr (PRE) {
#pragma unroll
      for (int p = 0; p < 4; ++p) *(v8bf*)&sB[p * 64 + ldr][ldc] = bs[p];
    } else {
#pragma unroll
      for (int p = 0; p < 4; ++p)
        *(v8bf*)&sB[p * 64 + ldr][ldc] = cvt8(bf0[p], bf1[p]);
    }
  };

  v8f acc[4] = {};
  load_stage(0);

  for (int c = 0; c < NCHUNK; ++c) {
    __syncthreads();
    store_stage();
    __syncthreads();
    if (c + 1 < NCHUNK) load_stage(c + 1);

#pragma unroll
    for (int ks = 0; ks < 2; ++ks) {
      const int ka = ks * 32;
      const int arow = tokTile * 16 + l16;
      v8bf alo = *(const v8bf*)&sA[arow][ka + lhalf * 8];
      v8bf ahi = *(const v8bf*)&sA[arow][ka + 16 + lhalf * 8];
      v16bf afrag = __builtin_shufflevector(alo, ahi,
          0,1,2,3,4,5,6,7,8,9,10,11,12,13,14,15);
#pragma unroll
      for (int t = 0; t < 4; ++t) {
        const int brow = exBase + t * 16 + l16;
        v8bf blo = *(const v8bf*)&sB[brow][ka + lhalf * 8];
        v8bf bhi = *(const v8bf*)&sB[brow][ka + 16 + lhalf * 8];
        v16bf bfrag = __builtin_shufflevector(blo, bhi,
            0,1,2,3,4,5,6,7,8,9,10,11,12,13,14,15);
        acc[t] = __builtin_amdgcn_wmma_f32_16x16x32_bf16(
            false, afrag, false, bfrag, (short)0, acc[t], false, false);
      }
    }
  }

  epilogue<32>(smem, acc, tokTile, exBase, l16, lhalf, tid, sBias,
               outIdx, outW, tok0);
}

extern "C" void kernel_launch(void* const* d_in, const int* in_sizes, int n_in,
                              void* d_out, int out_size, void* d_ws, size_t ws_size,
                              hipStream_t stream) {
  const float* hidden = (const float*)d_in[0];   // [T, 7168] f32
  const float* weight = (const float*)d_in[1];   // [256, 7168] f32
  const float* bias   = (const float*)d_in[2];   // [256] f32
  const int T = in_sizes[0] / HIDDEN;            // 8192

  int*   outIdx = (int*)d_out;                       // topk_idx as int32
  float* outW   = (float*)d_out + (size_t)T * TOPK;  // topk_w as f32

  dim3 block(NTHREADS), grid(T / TOKTILE);
  const size_t wbytes = (size_t)NEXP * HIDDEN * sizeof(bf16);
  if (ws_size >= wbytes) {
    bf16* wb = (bf16*)d_ws;
    convert_weight_kernel<<<dim3((NEXP * HIDDEN / 8) / 256), dim3(256), 0,
                            stream>>>(weight, wb);
#if HAVE_TDM
    deepseek_gate_kernel_tdm<<<grid, block, 0, stream>>>(hidden, wb, bias,
                                                         outIdx, outW);
#else
    deepseek_gate_kernel<true><<<grid, block, 0, stream>>>(
        hidden, nullptr, wb, bias, outIdx, outW);
#endif
  } else {
    deepseek_gate_kernel<false><<<grid, block, 0, stream>>>(
        hidden, weight, nullptr, bias, outIdx, outW);
  }
}
